// SlidingWindowAttention_35974646071444
// MI455X (gfx1250) — matrix operand order repaired and checked by reading, not compiled, not run on hardware
//
#include <hip/hip_runtime.h>

// ---------------------------------------------------------------------------
// Types for CDNA5 WMMA (wave32): D(16x16 f32) = A(16x32 f16) * B(32x16 f16) + C
// ---------------------------------------------------------------------------
typedef __attribute__((ext_vector_type(16))) _Float16 v16h;
typedef __attribute__((ext_vector_type(8)))  float    v8f;

union AB { uint4 u4[2]; v16h v; };

__device__ __forceinline__ v8f wmma_f16(v16h a, v16h b, v8f c) {
    return __builtin_amdgcn_wmma_f32_16x16x32_f16(false, a, false, b, (short)0, c,
                                                  false, false);
}

// max across the 16 lanes of each half-wave via ds_swizzle XOR patterns
__device__ __forceinline__ float halfmax16(float v) {
    v = fmaxf(v, __int_as_float(__builtin_amdgcn_ds_swizzle(__float_as_int(v), 0x041F)));
    v = fmaxf(v, __int_as_float(__builtin_amdgcn_ds_swizzle(__float_as_int(v), 0x081F)));
    v = fmaxf(v, __int_as_float(__builtin_amdgcn_ds_swizzle(__float_as_int(v), 0x101F)));
    v = fmaxf(v, __int_as_float(__builtin_amdgcn_ds_swizzle(__float_as_int(v), 0x201F)));
    return v;
}

// CDNA5 async global->LDS copy (16B per lane), tracked by ASYNCcnt
__device__ __forceinline__ void async_copy_b128(unsigned lds_addr, const void* gaddr) {
    asm volatile("global_load_async_to_lds_b128 %0, %1, off"
                 :: "v"(lds_addr), "v"((unsigned long long)(size_t)gaddr)
                 : "memory");
}
__device__ __forceinline__ void wait_asynccnt0() {
    asm volatile("s_wait_asynccnt 0x0" ::: "memory");
}

// ---------------------------------------------------------------------------
// Problem constants
// ---------------------------------------------------------------------------
#define T_SEQ   4096
#define D_MODEL 1024
#define N_HEADS 16
#define HEAD_D  64
#define WINDOW  256
#define LDQKV   3072
#define NEGB    (-1e30f)

// ---------------------------------------------------------------------------
// fp32 -> fp16 conversion, 8 elements/thread (b128 store)
// ---------------------------------------------------------------------------
__global__ void f32_to_f16(const float* __restrict__ src, _Float16* __restrict__ dst, int n) {
    int i = (blockIdx.x * blockDim.x + threadIdx.x) * 8;
    if (i < n) {
        float4 a = *(const float4*)(src + i);
        float4 b = *(const float4*)(src + i + 4);
        union { _Float16 h[8]; uint4 u; } pk;
        pk.h[0] = (_Float16)a.x; pk.h[1] = (_Float16)a.y;
        pk.h[2] = (_Float16)a.z; pk.h[3] = (_Float16)a.w;
        pk.h[4] = (_Float16)b.x; pk.h[5] = (_Float16)b.y;
        pk.h[6] = (_Float16)b.z; pk.h[7] = (_Float16)b.w;
        *(uint4*)(dst + i) = pk.u;
    }
}

// ---------------------------------------------------------------------------
// fp16 GEMM: C(M,N) = A(M,K) * B(K,N); N, K compile-time; double-buffered LDS.
// A tile staged via async global->LDS DMA (ASYNCcnt); B staged via registers
// with an element-scatter transpose (fragment layout requirement).
// Block tile 128x128, BK=32, 256 threads (8 waves); wave tile 32x64.
// ---------------------------------------------------------------------------
#define BM 128
#define BN 128
#define BK 32
#define LDSA 40
#define LDSB 40

template<int N, int K, bool F32OUT>
__global__ __launch_bounds__(256) void gemm_f16(
        const _Float16* __restrict__ A, const _Float16* __restrict__ B,
        _Float16* __restrict__ Cb, float* __restrict__ Cf) {
    __shared__ _Float16 As [2][BM * LDSA];
    __shared__ _Float16 Bts[2][BN * LDSB];   // transposed: Bts[n][k]

    const int tid  = threadIdx.x;
    const int wid  = tid >> 5;
    const int lane = tid & 31;
    const int hh   = lane >> 4;
    const int l15  = lane & 15;

    const int m0 = blockIdx.y * BM;
    const int n0 = blockIdx.x * BN;
    const int wm = (wid & 3) * 32;
    const int wn = (wid >> 2) * 64;

    const int ar   = tid >> 2;
    const int acc_ = (tid & 3) * 8;
    const int br   = tid >> 4;
    const int bcc  = (tid & 15) * 8;

    v8f acc[2][4];
#pragma unroll
    for (int i = 0; i < 2; ++i)
#pragma unroll
        for (int j = 0; j < 4; ++j)
#pragma unroll
            for (int r = 0; r < 8; ++r) acc[i][j][r] = 0.0f;

    uint4 bReg[2];

    auto issueAasync = [&](int kt, int buf) {
        const int k0 = kt * BK;
#pragma unroll
        for (int p = 0; p < 2; ++p) {
            unsigned lds = (unsigned)(size_t)(As[buf] + (ar + p * 64) * LDSA + acc_);
            async_copy_b128(lds, A + (size_t)(m0 + ar + p * 64) * K + k0 + acc_);
        }
    };
    auto loadBglobal = [&](int kt) {
        const int k0 = kt * BK;
#pragma unroll
        for (int p = 0; p < 2; ++p)
            bReg[p] = *(const uint4*)(B + (size_t)(k0 + br + p * 16) * N + n0 + bcc);
    };
    auto storeB = [&](int buf) {
#pragma unroll
        for (int p = 0; p < 2; ++p) {
            const _Float16* hb = (const _Float16*)&bReg[p];
#pragma unroll
            for (int j = 0; j < 8; ++j)
                Bts[buf][(bcc + j) * LDSB + (br + p * 16)] = hb[j];
        }
    };

    constexpr int NT = K / BK;
    issueAasync(0, 0);
    loadBglobal(0);
    storeB(0);
    wait_asynccnt0();
    __syncthreads();

    for (int kt = 0; kt < NT; ++kt) {
        const int cur = kt & 1;
        if (kt + 1 < NT) {
            issueAasync(kt + 1, cur ^ 1);   // DMA next A tile under WMMAs
            loadBglobal(kt + 1);
        }

        AB afrag[2], bfrag[4];
#pragma unroll
        for (int i = 0; i < 2; ++i) {
            int row = wm + i * 16 + l15;
            afrag[i].u4[0] = *(const uint4*)(As[cur] + row * LDSA + hh * 8);
            afrag[i].u4[1] = *(const uint4*)(As[cur] + row * LDSA + 16 + hh * 8);
        }
#pragma unroll
        for (int j = 0; j < 4; ++j) {
            int col = wn + j * 16 + l15;
            bfrag[j].u4[0] = *(const uint4*)(Bts[cur] + col * LDSB + hh * 8);
            bfrag[j].u4[1] = *(const uint4*)(Bts[cur] + col * LDSB + 16 + hh * 8);
        }
#pragma unroll
        for (int i = 0; i < 2; ++i)
#pragma unroll
            for (int j = 0; j < 4; ++j)
                acc[i][j] = wmma_f16(afrag[i].v, bfrag[j].v, acc[i][j]);

        if (kt + 1 < NT) storeB(cur ^ 1);
        wait_asynccnt0();                    // drain async writes into nxt
        __syncthreads();                     // publish nxt / retire cur
    }

#pragma unroll
    for (int i = 0; i < 2; ++i)
#pragma unroll
        for (int j = 0; j < 4; ++j) {
            const int row0 = m0 + wm + i * 16 + 8 * hh;
            const int col  = n0 + wn + j * 16 + l15;
            if constexpr (F32OUT) {
                float* p = Cf + (size_t)row0 * N + col;
#pragma unroll
                for (int r = 0; r < 8; ++r) p[r * N] = acc[i][j][r];
            } else {
                _Float16* p = Cb + (size_t)row0 * N + col;
#pragma unroll
                for (int r = 0; r < 8; ++r) p[r * N] = (_Float16)acc[i][j][r];
            }
        }
}

// ---------------------------------------------------------------------------
// Transpose V slice of qkv: vt[h*64+d][t] = qkv[t][2048 + h*64 + d]
// ---------------------------------------------------------------------------
__global__ void transpose_v(const _Float16* __restrict__ qkv, _Float16* __restrict__ vt) {
    __shared__ _Float16 s[32][33];
    int t0 = blockIdx.x * 32;
    int c0 = blockIdx.y * 32;
    int x = threadIdx.x, y = threadIdx.y;   // 32 x 8
    for (int yy = y; yy < 32; yy += 8)
        s[yy][x] = qkv[(size_t)(t0 + yy) * LDQKV + 2048 + c0 + x];
    __syncthreads();
    for (int yy = y; yy < 32; yy += 8)
        vt[(size_t)(c0 + yy) * T_SEQ + t0 + x] = s[x][yy];
}

// ---------------------------------------------------------------------------
// Sliding-window flash attention: one wave per (head, 16-query tile).
// 9 key tiles of 32: QK^T 4 WMMAs, PV 4 WMMAs, row-sum 1 WMMA (P x ones).
// ---------------------------------------------------------------------------
__global__ __launch_bounds__(128) void attn_kernel(
        const _Float16* __restrict__ qkv, const _Float16* __restrict__ vt,
        _Float16* __restrict__ attnb) {
    const int wid  = threadIdx.x >> 5;
    const int lane = threadIdx.x & 31;
    const int hh   = lane >> 4;
    const int l15  = lane & 15;

    const int unit = blockIdx.x * 4 + wid;  // 4096 units
    const int head = unit & 15;
    const int q0   = (unit >> 4) * 16;

    __shared__ _Float16 pshare[4][16 * 40];
    _Float16* ps = &pshare[wid][0];

    const int qcol = head * HEAD_D;
    const int kcol = 1024 + head * HEAD_D;

    // Q A-fragments, pre-scaled by 1/sqrt(64) (exact power of two in fp16)
    AB qa[2];
    {
        const _Float16* qp = qkv + (size_t)(q0 + l15) * LDQKV + qcol;
#pragma unroll
        for (int i = 0; i < 2; ++i) {
            int db = i * 32 + hh * 16;
            qa[i].u4[0] = *(const uint4*)(qp + db);
            qa[i].u4[1] = *(const uint4*)(qp + db + 8);
            qa[i].v = qa[i].v * (_Float16)0.125f;
        }
    }

    // ones B-fragment for row-sum WMMA (0x3C00 = 1.0h)
    AB onesf;
    onesf.u4[0] = make_uint4(0x3C003C00u, 0x3C003C00u, 0x3C003C00u, 0x3C003C00u);
    onesf.u4[1] = onesf.u4[0];

    v8f o[4], os;
    float mrow[8];
#pragma unroll
    for (int n = 0; n < 4; ++n)
#pragma unroll
        for (int r = 0; r < 8; ++r) o[n][r] = 0.0f;
#pragma unroll
    for (int r = 0; r < 8; ++r) { os[r] = 0.0f; mrow[r] = NEGB; }

    const bool edge = (q0 < WINDOW);

    for (int kt = 0; kt < 9; ++kt) {
        const int kt0 = q0 - WINDOW + kt * 32;
        if (kt0 + 31 < 0) continue;

        // ---- S = (Q/8) K^T ----
        v8f s[2];
#pragma unroll
        for (int j = 0; j < 2; ++j)
#pragma unroll
            for (int r = 0; r < 8; ++r) s[j][r] = 0.0f;

#pragma unroll
        for (int j = 0; j < 2; ++j) {
            int key = kt0 + j * 16 + l15;
            if (key < 0) key = 0;
            const _Float16* kp = qkv + (size_t)key * LDQKV + kcol + hh * 16;
            AB kb0, kb1;
            kb0.u4[0] = *(const uint4*)(kp);
            kb0.u4[1] = *(const uint4*)(kp + 8);
            kb1.u4[0] = *(const uint4*)(kp + 32);
            kb1.u4[1] = *(const uint4*)(kp + 40);
            s[j] = wmma_f16(qa[0].v, kb0.v, s[j]);
            s[j] = wmma_f16(qa[1].v, kb1.v, s[j]);
        }

        // ---- mask only the window-edge / diagonal / sequence-start tiles ----
        if (edge || kt == 0 || kt == 8) {
#pragma unroll
            for (int j = 0; j < 2; ++j)
#pragma unroll
                for (int r = 0; r < 8; ++r) {
                    int q   = q0 + r + 8 * hh;
                    int key = kt0 + j * 16 + l15;
                    bool ok = (key >= 0) && (key <= q) && (key >= q - (WINDOW - 1));
                    s[j][r] = ok ? s[j][r] : NEGB;
                }
        }

        // ---- online softmax: row max via ds_swizzle, row sum via WMMA ----
        float alpha[8];
#pragma unroll
        for (int r = 0; r < 8; ++r) {
            float t = halfmax16(fmaxf(s[0][r], s[1][r]));
            float mn = fmaxf(mrow[r], t);
            alpha[r] = __expf(mrow[r] - mn);
            mrow[r]  = mn;
        }
#pragma unroll
        for (int j = 0; j < 2; ++j)
#pragma unroll
            for (int r = 0; r < 8; ++r)
                s[j][r] = __expf(s[j][r] - mrow[r]);
#pragma unroll
        for (int n = 0; n < 4; ++n)
#pragma unroll
            for (int r = 0; r < 8; ++r) o[n][r] *= alpha[r];
#pragma unroll
        for (int r = 0; r < 8; ++r) os[r] *= alpha[r];

        // ---- P: C-layout -> A-layout via per-wave LDS staging ----
#pragma unroll
        for (int j = 0; j < 2; ++j)
#pragma unroll
            for (int r = 0; r < 8; ++r)
                ps[(r + 8 * hh) * 40 + j * 16 + l15] = (_Float16)s[j][r];
        __builtin_amdgcn_wave_barrier();
        AB pa;
        pa.u4[0] = *(const uint4*)(ps + l15 * 40 + hh * 16);
        pa.u4[1] = *(const uint4*)(ps + l15 * 40 + hh * 16 + 8);
        __builtin_amdgcn_wave_barrier();

        // ---- row sums: os += P x ones ----
        os = wmma_f16(pa.v, onesf.v, os);

        // ---- O += P * V ----
        int tb = kt0 + hh * 16;
        if (tb < 0) tb = 0;
#pragma unroll
        for (int n = 0; n < 4; ++n) {
            int d = n * 16 + l15;
            const _Float16* vp = vt + (size_t)(head * HEAD_D + d) * T_SEQ + tb;
            AB vb;
            vb.u4[0] = *(const uint4*)(vp);
            vb.u4[1] = *(const uint4*)(vp + 8);
            o[n] = wmma_f16(pa.v, vb.v, o[n]);
        }
    }

#pragma unroll
    for (int n = 0; n < 4; ++n)
#pragma unroll
        for (int r = 0; r < 8; ++r) {
            int row = q0 + r + 8 * hh;
            int col = head * HEAD_D + n * 16 + l15;
            attnb[(size_t)row * D_MODEL + col] = (_Float16)(o[n][r] / os[r]);
        }
}

// ---------------------------------------------------------------------------
// Launch
// ---------------------------------------------------------------------------
extern "C" void kernel_launch(void* const* d_in, const int* in_sizes, int n_in,
                              void* d_out, int out_size, void* d_ws, size_t ws_size,
                              hipStream_t stream) {
    const float* x     = (const float*)d_in[0];   // (4096, 1024)
    const float* w_qkv = (const float*)d_in[1];   // (1024, 3072)
    const float* w_out = (const float*)d_in[2];   // (1024, 1024)
    float* out = (float*)d_out;                   // (4096, 1024)

    _Float16* xb    = (_Float16*)d_ws;
    _Float16* wqkvb = xb    + (size_t)T_SEQ * D_MODEL;
    _Float16* woutb = wqkvb + (size_t)D_MODEL * LDQKV;
    _Float16* qkvb  = woutb + (size_t)D_MODEL * D_MODEL;
    _Float16* vtb   = qkvb  + (size_t)T_SEQ * LDQKV;
    _Float16* attnb = vtb   + (size_t)D_MODEL * T_SEQ;

    const int n1 = T_SEQ * D_MODEL, n2 = D_MODEL * LDQKV, n3 = D_MODEL * D_MODEL;
    f32_to_f16<<<(n1 / 8 + 255) / 256, 256, 0, stream>>>(x, xb, n1);
    f32_to_f16<<<(n2 / 8 + 255) / 256, 256, 0, stream>>>(w_qkv, wqkvb, n2);
    f32_to_f16<<<(n3 / 8 + 255) / 256, 256, 0, stream>>>(w_out, woutb, n3);

    gemm_f16<LDQKV, D_MODEL, false><<<dim3(LDQKV / BN, T_SEQ / BM), 256, 0, stream>>>(
        xb, wqkvb, qkvb, nullptr);

    transpose_v<<<dim3(T_SEQ / 32, D_MODEL / 32), dim3(32, 8), 0, stream>>>(qkvb, vtb);

    attn_kernel<<<(N_HEADS * (T_SEQ / 16)) / 4, 128, 0, stream>>>(qkvb, vtb, attnb);

    gemm_f16<D_MODEL, D_MODEL, true><<<dim3(D_MODEL / BN, T_SEQ / BM), 256, 0, stream>>>(
        attnb, woutb, nullptr, out);
}
